// AttentionalPooler_11398843204166
// MI455X (gfx1250) — compile-verified, matrix operand order checked by reading, not run on hardware
//
#include <hip/hip_runtime.h>
#include <float.h>

// Problem constants (match reference setup_inputs)
#define B_   32
#define C_   1280
#define HW_  1024
#define TD_  768
#define LD_  64
#define EPS_ 1e-5f
#define LSCALE_ 0.125f   // 64^-0.5

typedef __bf16 bf16_t;
typedef bf16_t v8bf  __attribute__((ext_vector_type(8)));
typedef bf16_t v16bf __attribute__((ext_vector_type(16)));
typedef float  v8f   __attribute__((ext_vector_type(8)));

// ---------------- wave32 / block reductions ----------------
__device__ __forceinline__ float wsum(float v) {
#pragma unroll
    for (int o = 16; o > 0; o >>= 1) v += __shfl_xor(v, o, 32);
    return v;
}
__device__ __forceinline__ float wmaxf(float v) {
#pragma unroll
    for (int o = 16; o > 0; o >>= 1) v = fmaxf(v, __shfl_xor(v, o, 32));
    return v;
}
__device__ __forceinline__ float block_sum(float v, float* red) {
    int lane = threadIdx.x & 31, wid = threadIdx.x >> 5;
    v = wsum(v);
    if (lane == 0) red[wid] = v;
    __syncthreads();
    int nw = (blockDim.x + 31) >> 5;
    float r = (threadIdx.x < (unsigned)nw) ? red[threadIdx.x] : 0.f;
    if (wid == 0) { r = wsum(r); if (lane == 0) red[0] = r; }
    __syncthreads();
    float out = red[0];
    __syncthreads();
    return out;
}
__device__ __forceinline__ float block_max(float v, float* red) {
    int lane = threadIdx.x & 31, wid = threadIdx.x >> 5;
    v = wmaxf(v);
    if (lane == 0) red[wid] = v;
    __syncthreads();
    int nw = (blockDim.x + 31) >> 5;
    float r = (threadIdx.x < (unsigned)nw) ? red[threadIdx.x] : -FLT_MAX;
    if (wid == 0) { r = wmaxf(r); if (lane == 0) red[0] = r; }
    __syncthreads();
    float out = red[0];
    __syncthreads();
    return out;
}

// ---------------- K1: q-path (tiny, one workgroup) ----------------
// lora_q[d] = LN( LN(tok)*g+b @ to_q_w^T ) @ lora_q_w^T
__global__ void k_prep(const float* __restrict__ tok, const float* __restrict__ toqw,
                       const float* __restrict__ lnqg, const float* __restrict__ lnqb,
                       const float* __restrict__ lqw, float* __restrict__ lora_q)
{
    __shared__ float red[32];
    __shared__ float qn[TD_];
    __shared__ float buf[C_];
    int tid = threadIdx.x;

    float s1 = 0.f, s2 = 0.f;
    for (int t = tid; t < TD_; t += 256) { float v = tok[t]; s1 += v; s2 = fmaf(v, v, s2); }
    s1 = block_sum(s1, red); s2 = block_sum(s2, red);
    float mu = s1 * (1.f / TD_);
    float rs = rsqrtf(s2 * (1.f / TD_) - mu * mu + EPS_);
    for (int t = tid; t < TD_; t += 256)
        qn[t] = fmaf(lnqg[t], (tok[t] - mu) * rs, lnqb[t]);
    __syncthreads();

    for (int c = tid; c < C_; c += 256) {
        float acc = 0.f;
        const float* w = toqw + (size_t)c * TD_;
        for (int t = 0; t < TD_; t++) acc = fmaf(qn[t], w[t], acc);
        buf[c] = acc;
    }
    __syncthreads();

    s1 = 0.f; s2 = 0.f;
    for (int c = tid; c < C_; c += 256) { float v = buf[c]; s1 += v; s2 = fmaf(v, v, s2); }
    s1 = block_sum(s1, red); s2 = block_sum(s2, red);
    mu = s1 * (1.f / C_);
    rs = rsqrtf(s2 * (1.f / C_) - mu * mu + EPS_);

    if (tid < LD_) {
        float acc = 0.f;
        const float* w = lqw + (size_t)tid * C_;
        for (int c = 0; c < C_; c++) acc = fmaf((buf[c] - mu) * rs, w[c], acc);
        lora_q[tid] = acc;
    }
}

// ---------------- K1b: lora_k_w -> fragment-ready bf16 B layout ----------------
// Bfrag[kb][nt][lane][16]: lane<16 holds col d=nt*16+lane, k = kb*32 + {0..7,16..23};
//                          lane>=16 same col-group, k = kb*32 + {8..15,24..31}.
__global__ void k_bfrag(const float* __restrict__ lkw, bf16_t* __restrict__ Bfrag)
{
    int idx = blockIdx.x * 256 + threadIdx.x;
    if (idx >= 40 * 4 * 32 * 16) return;
    int e    = idx & 15;
    int lane = (idx >> 4) & 31;
    int nt   = (idx >> 9) & 3;
    int kb   = idx >> 11;
    int l15 = lane & 15, half = lane >> 4;
    int d  = nt * 16 + l15;
    int kl = e + ((e < 8) ? 0 : 8) + half * 8;
    int c  = kb * 32 + kl;
    Bfrag[idx] = (bf16_t)lkw[(size_t)d * C_ + c];
}

// ---------------- K4: per-position stats of layer_infeat (fills L2) ----------------
__global__ void k_xstats(const float* __restrict__ xf, float* __restrict__ mu,
                         float* __restrict__ rsig)
{
    int b = blockIdx.x, j = threadIdx.x;
    const float* p = xf + (size_t)b * C_ * HW_ + j;
    float s1 = 0.f, s2 = 0.f;
#pragma unroll 4
    for (int c = 0; c < C_; c++) {
        float v = p[(size_t)c * HW_];      // coalesced across j
        s1 += v; s2 = fmaf(v, v, s2);
    }
    float m = s1 * (1.f / C_);
    mu[b * HW_ + j]   = m;
    rsig[b * HW_ + j] = rsqrtf(s2 * (1.f / C_) - m * m + EPS_);
}

// ---------------- K2: double-LN + WMMA lora_k GEMM + sim ----------------
// One block = 16 rows of one batch. 16 waves: wave-per-row LayerNorm chain.
// Then waves 0..3 run v_wmma_f32_16x16x32_bf16, K=1280 in 40 steps.
__global__ __launch_bounds__(512, 1) void k_sim(
    const float* __restrict__ kin, const float* __restrict__ lnkg,
    const float* __restrict__ lnkb, const bf16_t* __restrict__ Bfrag,
    const float* __restrict__ lora_q, const float* __restrict__ maskf,
    float* __restrict__ sim)
{
    __shared__ __align__(16) bf16_t Als[16 * C_];   // 40 KB normalized A tile
    __shared__ float simp[16];
    const int b    = blockIdx.x >> 6;       // 64 row-blocks per batch
    const int j0   = (blockIdx.x & 63) * 16;
    const int wave = threadIdx.x >> 5;
    const int lane = threadIdx.x & 31;

    {   // ---- LN(LN(row)*g+b), one wave per row, NT loads (stream; keep infeat in L2)
        const int j = j0 + wave;
        const float* row = kin + ((size_t)b * HW_ + j) * C_;
        float kv[40];
        float s1 = 0.f, s2 = 0.f;
#pragma unroll
        for (int i = 0; i < 40; i++) {
            float v = __builtin_nontemporal_load(row + lane + 32 * i);
            kv[i] = v; s1 += v; s2 = fmaf(v, v, s2);
        }
        s1 = wsum(s1); s2 = wsum(s2);
        float mu = s1 * (1.f / C_);
        float rs = rsqrtf(s2 * (1.f / C_) - mu * mu + EPS_);
        float t1 = 0.f, t2 = 0.f;
#pragma unroll
        for (int i = 0; i < 40; i++) {
            int c = lane + 32 * i;
            float kk = fmaf(lnkg[c], (kv[i] - mu) * rs, lnkb[c]);
            kv[i] = kk; t1 += kk; t2 = fmaf(kk, kk, t2);
        }
        t1 = wsum(t1); t2 = wsum(t2);
        float mu2 = t1 * (1.f / C_);
        float rs2 = rsqrtf(t2 * (1.f / C_) - mu2 * mu2 + EPS_);
#pragma unroll
        for (int i = 0; i < 40; i++)
            Als[wave * C_ + lane + 32 * i] = (bf16_t)((kv[i] - mu2) * rs2);
    }
    if (threadIdx.x < 16) simp[threadIdx.x] = 0.f;
    __syncthreads();

    if (wave < 4) {
        const int nt = wave, l15 = lane & 15, half = lane >> 4;
        v8f acc = {};
        for (int kb = 0; kb < 40; kb++) {
            // A fragment: row = l15, k-halves per ISA 16-bit A layout
            v8bf alo = *(const v8bf*)&Als[l15 * C_ + kb * 32 + half * 8];
            v8bf ahi = *(const v8bf*)&Als[l15 * C_ + kb * 32 + half * 8 + 16];
            // B fragment: contiguous 32B per lane from pre-swizzled buffer
            const v8bf* bp = (const v8bf*)&Bfrag[(((size_t)kb * 4 + nt) * 32 + lane) * 16];
            v8bf blo = bp[0], bhi = bp[1];
            v16bf a, bm;
#pragma unroll
            for (int i = 0; i < 8; i++) { a[i] = alo[i]; a[i + 8] = ahi[i];
                                          bm[i] = blo[i]; bm[i + 8] = bhi[i]; }
            acc = __builtin_amdgcn_wmma_f32_16x16x32_bf16(
                      false, a, false, bm, (short)0, acc, false, false);
        }
        // D layout: VGPR r -> M = r + 8*half, N = nt*16 + l15. Contract with lora_q.
        float lq = lora_q[nt * 16 + l15];
#pragma unroll
        for (int r = 0; r < 8; r++)
            atomicAdd(&simp[r + half * 8], acc[r] * lq);
    }
    __syncthreads();

    if (threadIdx.x < 16) {
        int j = j0 + threadIdx.x;
        float m = maskf[b * HW_ + j];
        float s = simp[threadIdx.x] * LSCALE_;
        sim[b * HW_ + j] = (m > 0.f) ? s : -FLT_MAX;
    }
}

// ---------------- K3: softmax + fold stats into j-weights ----------------
__global__ void k_soft(const float* __restrict__ sim, const float* __restrict__ mu,
                       const float* __restrict__ rsig, float* __restrict__ wA,
                       float* __restrict__ wB, float* __restrict__ cA,
                       float* __restrict__ cB)
{
    __shared__ float red[32];
    int b = blockIdx.x, j = threadIdx.x;
    float s  = sim[b * HW_ + j];
    float mx = block_max(s, red);
    float e  = __expf(s - mx);
    float dn = block_sum(e, red);
    float attn = e / dn;
    float m = mu[b * HW_ + j], r = rsig[b * HW_ + j];
    wA[b * HW_ + j] = attn * r;
    wB[b * HW_ + j] = r * (1.f / HW_);
    float a1 = block_sum(attn * m * r, red);
    float a2 = block_sum(m * r, red);
    if (j == 0) { cA[b] = a1; cB[b] = a2 * (1.f / HW_); }
}

// ---------------- K5: weighted j-sums over layer_infeat (L2-resident) -> out ----
__global__ void k_out(const float* __restrict__ xf, const float* __restrict__ wA,
                      const float* __restrict__ wB, const float* __restrict__ cA,
                      const float* __restrict__ cB, const float* __restrict__ gx,
                      const float* __restrict__ bx, float* __restrict__ out)
{
    int b = blockIdx.x / 5;
    int c = (blockIdx.x % 5) * 256 + threadIdx.x;
    const float4* xp = (const float4*)(xf + ((size_t)b * C_ + c) * HW_);
    const float4* wa = (const float4*)(wA + b * HW_);   // uniform -> scalar loads
    const float4* wb = (const float4*)(wB + b * HW_);
    float accA = 0.f, accB = 0.f;
#pragma unroll 2
    for (int q = 0; q < HW_ / 4; q++) {
        float4 xv = xp[q]; float4 a = wa[q]; float4 w = wb[q];
        accA = fmaf(a.x, xv.x, fmaf(a.y, xv.y, fmaf(a.z, xv.z, fmaf(a.w, xv.w, accA))));
        accB = fmaf(w.x, xv.x, fmaf(w.y, xv.y, fmaf(w.z, xv.z, fmaf(w.w, xv.w, accB))));
    }
    float g = gx[c], be = bx[c];
    float fgv = fmaf(g, accA - cA[b], be);
    float bgv = g * (accB - cB[b] - (accA - cA[b]));
    out[(size_t)b * 2 * C_ + c]        = fgv;
    out[(size_t)b * 2 * C_ + C_ + c]   = bgv;
}

extern "C" void kernel_launch(void* const* d_in, const int* in_sizes, int n_in,
                              void* d_out, int out_size, void* d_ws, size_t ws_size,
                              hipStream_t stream)
{
    const float* infeat  = (const float*)d_in[0];
    const float* inquery = (const float*)d_in[1];
    const float* tok     = (const float*)d_in[2];
    const float* toqw    = (const float*)d_in[3];
    const float* lnxg    = (const float*)d_in[4];
    const float* lnxb    = (const float*)d_in[5];
    const float* lnkg    = (const float*)d_in[6];
    const float* lnkb    = (const float*)d_in[7];
    const float* lnqg    = (const float*)d_in[8];
    const float* lnqb    = (const float*)d_in[9];
    const float* lqw     = (const float*)d_in[10];
    const float* lkw     = (const float*)d_in[11];
    const float* maskf   = (const float*)d_in[12];

    char* ws = (char*)d_ws;
    float*  lora_q = (float*)(ws + 0);                // 64 f32
    bf16_t* Bfrag  = (bf16_t*)(ws + 256);             // 81920 bf16 = 160 KB
    float*  sim    = (float*)(ws + 164096);           // 32*1024 f32
    float*  wA     = (float*)(ws + 295168);
    float*  wB     = (float*)(ws + 426240);
    float*  muA    = (float*)(ws + 557312);
    float*  rsigA  = (float*)(ws + 688384);
    float*  cA     = (float*)(ws + 819456);
    float*  cB     = (float*)(ws + 819584);

    k_prep  <<<1, 256, 0, stream>>>(tok, toqw, lnqg, lnqb, lqw, lora_q);
    k_bfrag <<<(40 * 4 * 32 * 16 + 255) / 256, 256, 0, stream>>>(lkw, Bfrag);
    // infeat pass 1 first so it is resident in the 192MB L2 ...
    k_xstats<<<B_, HW_, 0, stream>>>(infeat, muA, rsigA);
    // ... then stream inquery non-temporally (does not evict infeat)
    k_sim   <<<B_ * (HW_ / 16), 512, 0, stream>>>(inquery, lnkg, lnkb, Bfrag,
                                                  lora_q, maskf, sim);
    k_soft  <<<B_, HW_, 0, stream>>>(sim, muA, rsigA, wA, wB, cA, cB);
    k_out   <<<B_ * (C_ / 256), 256, 0, stream>>>(infeat, wA, wB, cA, cB,
                                                  lnxg, lnxb, (float*)d_out);
}
